// DescriptorNetwork_89541478187667
// MI455X (gfx1250) — matrix operand order repaired and checked by reading, not compiled, not run on hardware
//
#include <hip/hip_runtime.h>

// ---------------- CDNA5 WMMA types ----------------
typedef __attribute__((ext_vector_type(16))) __bf16 v16bf;
typedef __attribute__((ext_vector_type(8)))  float  v8f;
typedef __attribute__((ext_vector_type(4)))  unsigned int u32x4;

union Frag { v16bf v; u32x4 u[2]; };

__device__ __forceinline__ unsigned short bf16rne(float f) {
  unsigned int u = __float_as_uint(f);
  unsigned int r = u + 0x7FFFu + ((u >> 16) & 1u);   // round-to-nearest-even
  return (unsigned short)(r >> 16);
}
__device__ __forceinline__ float lrelu(float v) { return v > 0.f ? v : 0.01f * v; }

__device__ __forceinline__ v8f wmma_bf16(const Frag& a, const Frag& b, v8f c) {
  return __builtin_amdgcn_wmma_f32_16x16x32_bf16(false, a.v, false, b.v, (short)0, c, false, false);
}

// A-matrix 16x32 bf16 fragment from a row-major tile (rowbase = this lane's row).
// Layout (matches packed B): lanes 0-15 hold K {0..15}, lanes 16-31 hold K {16..31}.
__device__ __forceinline__ Frag load_a_frag(const unsigned short* rowbase, int s, int lane) {
  const int kb = (lane < 16) ? 0 : 16;
  const unsigned short* p = rowbase + s * 32 + kb;
  Frag f;
  f.u[0] = *(const u32x4*)(p);        // K = kb .. kb+7
  f.u[1] = *(const u32x4*)(p + 8);    // K = kb+8 .. kb+15
  return f;
}
// B-matrix 32x16 bf16 fragment, pre-swizzled in memory: 32 lanes x 16 bf16 contiguous.
__device__ __forceinline__ Frag load_b_frag(const unsigned short* fragbase, int lane) {
  const unsigned short* p = fragbase + lane * 16;
  Frag f;
  f.u[0] = *(const u32x4*)(p);
  f.u[1] = *(const u32x4*)(p + 8);
  return f;
}

// ---------------- problem dims ----------------
#define CRY   8192
#define NNODE (CRY*4)

// ---------------- embed: x = elem_fea @ Wemb + b  ||  elem_weights ----------------
__global__ void __launch_bounds__(256) k_embed(
    const float* __restrict__ ef, const float* __restrict__ ewt,
    const float* __restrict__ W, const float* __restrict__ b,
    float* __restrict__ X, unsigned short* __restrict__ XB) {
  __shared__ float row[200];
  const int n = blockIdx.x, c = threadIdx.x;
  for (int k = c; k < 200; k += 256) row[k] = ef[n * 200 + k];
  __syncthreads();
  float v;
  if (c < 255) {
    v = b[c];
    for (int k = 0; k < 200; ++k) v = fmaf(row[k], W[k * 255 + c], v);
  } else {
    v = ewt[n];
  }
  X[n * 256 + c] = v;
  XB[n * 256 + c] = bf16rne(v);
}

// ---------------- weight repack (f32 -> pre-swizzled bf16 B-fragments) ----------------
struct PackSrc { const float* gW[3]; const float* mW[3]; const float* m2W[3]; };

// graph fc1: per head build [256 x 2048] = [gate_top | gate_bot | msg_top | msg_bot]
// frag index: head h, Ntile t (0..127), Kstep s (0..7); frag = 512 ushort
__global__ void __launch_bounds__(256) k_pack_graph_fc1(PackSrc ps, unsigned short* __restrict__ dst) {
  const int tid = blockIdx.x * 256 + threadIdx.x;
  const int lane = tid & 31, s = (tid >> 5) & 7, t = (tid >> 8) & 127, h = tid >> 15;
  if (h >= 3) return;
  const int col = t * 16 + (lane & 15);
  const int kb  = s * 32 + ((lane & 16) ? 16 : 0);
  const float* src; int r0, cc;
  if      (col <  512) { src = ps.gW[h]; r0 = 0;   cc = col;        }
  else if (col < 1024) { src = ps.gW[h]; r0 = 256; cc = col - 512;  }
  else if (col < 1536) { src = ps.mW[h]; r0 = 0;   cc = col - 1024; }
  else                 { src = ps.mW[h]; r0 = 256; cc = col - 1536; }
  unsigned int w[8];
  for (int e2 = 0; e2 < 8; ++e2) {
    const int k = kb + 2 * e2;
    unsigned int lo = bf16rne(src[(r0 + k)     * 512 + cc]);
    unsigned int hi = bf16rne(src[(r0 + k + 1) * 512 + cc]);
    w[e2] = lo | (hi << 16);
  }
  unsigned short* d = dst + (((size_t)(h * 128 + t) * 8 + s) * 32 + lane) * 16;
  u32x4 lo4 = {w[0], w[1], w[2], w[3]}, hi4 = {w[4], w[5], w[6], w[7]};
  ((u32x4*)d)[0] = lo4; ((u32x4*)d)[1] = hi4;
}

// msg.out.W (512x256) pack: head h, Ntile nt (0..15), Kstep s (0..15)
__global__ void __launch_bounds__(256) k_pack_m2(PackSrc ps, unsigned short* __restrict__ dst) {
  const int tid = blockIdx.x * 256 + threadIdx.x;
  const int lane = tid & 31, s = (tid >> 5) & 15, nt = (tid >> 9) & 15, h = tid >> 13;
  if (h >= 3) return;
  const float* W = ps.m2W[h];
  const int col = nt * 16 + (lane & 15);
  const int kb  = s * 32 + ((lane & 16) ? 16 : 0);
  unsigned int w[8];
  for (int e2 = 0; e2 < 8; ++e2) {
    const int k = kb + 2 * e2;
    unsigned int lo = bf16rne(W[k * 256 + col]);
    unsigned int hi = bf16rne(W[(k + 1) * 256 + col]);
    w[e2] = lo | (hi << 16);
  }
  unsigned short* d = dst + (((size_t)(h * 16 + nt) * 16 + s) * 32 + lane) * 16;
  u32x4 lo4 = {w[0], w[1], w[2], w[3]}, hi4 = {w[4], w[5], w[6], w[7]};
  ((u32x4*)d)[0] = lo4; ((u32x4*)d)[1] = hi4;
}

// cry fc1: per head [256 x 1024] = [gate_fc1 | msg_fc1]; t 0..63, s 0..7
__global__ void __launch_bounds__(256) k_pack_cry_fc1(PackSrc ps, unsigned short* __restrict__ dst) {
  const int tid = blockIdx.x * 256 + threadIdx.x;
  const int lane = tid & 31, s = (tid >> 5) & 7, t = (tid >> 8) & 63, h = tid >> 14;
  if (h >= 3) return;
  const int col = t * 16 + (lane & 15);
  const int kb  = s * 32 + ((lane & 16) ? 16 : 0);
  const float* src; int cc;
  if (col < 512) { src = ps.gW[h]; cc = col; } else { src = ps.mW[h]; cc = col - 512; }
  unsigned int w[8];
  for (int e2 = 0; e2 < 8; ++e2) {
    const int k = kb + 2 * e2;
    unsigned int lo = bf16rne(src[k * 512 + cc]);
    unsigned int hi = bf16rne(src[(k + 1) * 512 + cc]);
    w[e2] = lo | (hi << 16);
  }
  unsigned short* d = dst + (((size_t)(h * 64 + t) * 8 + s) * 32 + lane) * 16;
  u32x4 lo4 = {w[0], w[1], w[2], w[3]}, hi4 = {w[4], w[5], w[6], w[7]};
  ((u32x4*)d)[0] = lo4; ((u32x4*)d)[1] = hi4;
}

// ---------------- fused graph layer ----------------
struct HeadSmall { const float *gb1, *gw2, *gb2, *mb1, *mb2, *pw; };
struct HeadsArg  { HeadSmall h[3]; };

// LDS byte offsets (dynamic LDS, ~225 KB of the 320 KB WGP LDS)
#define L_F     0        // float[16*2048] fc1 out; first 64KB aliased as MO float[64*256]
#define L_XT    131072   // ushort[16*256] bf16 x tile
#define L_H     139264   // ushort[64*512] bf16 edge hidden
#define L_POOL  204800   // float[16*256] pooled accumulator
#define L_GB1   221184
#define L_GW2   223232
#define L_MB1   225280
#define L_MB2   227328
#define L_PART  228352   // float[64*4]
#define L_GATE  229376   // float[64]
#define L_COEF  229632   // float[64]
#define L_TOTAL 229888

__global__ void __launch_bounds__(256) k_graph_layer(
    float* __restrict__ X, unsigned short* __restrict__ XB,
    const unsigned short* __restrict__ wpack,   // [3][128][8] B-frags
    const unsigned short* __restrict__ wm2,     // [3][16][16]  B-frags
    const float* __restrict__ ewt, HeadsArg ha) {
  extern __shared__ char smem[];
  float*          F    = (float*)(smem + L_F);
  float*          MO   = (float*)(smem + L_F);      // aliases F (used after F consumed)
  unsigned short* xt   = (unsigned short*)(smem + L_XT);
  unsigned short* H    = (unsigned short*)(smem + L_H);
  float*          pool = (float*)(smem + L_POOL);
  float*          gb1s = (float*)(smem + L_GB1);
  float*          gw2s = (float*)(smem + L_GW2);
  float*          mb1s = (float*)(smem + L_MB1);
  float*          mb2s = (float*)(smem + L_MB2);
  float*          part = (float*)(smem + L_PART);
  float*          gate = (float*)(smem + L_GATE);
  float*          coef = (float*)(smem + L_COEF);

  const int tid = threadIdx.x, lane = tid & 31, wave = tid >> 5;
  const int nodeBase = blockIdx.x * 16;

  for (int i = tid; i < 16 * 256; i += 256) {
    xt[i]   = XB[nodeBase * 256 + i];
    pool[i] = 0.f;
  }
  __syncthreads();

  for (int hh = 0; hh < 3; ++hh) {
    const HeadSmall hp = ha.h[hh];
    for (int i = tid; i < 512; i += 256) { gb1s[i] = hp.gb1[i]; gw2s[i] = hp.gw2[i]; mb1s[i] = hp.mb1[i]; }
    if (tid < 256) mb2s[tid] = hp.mb2[tid];
    __syncthreads();

    // ---- fc1: [16x256] @ [256x2048] -> F (A,B parts of gate & msg MLPs) ----
    {
      Frag ax[8];
      const unsigned short* rowb = xt + (lane & 15) * 256;
      #pragma unroll
      for (int s = 0; s < 8; ++s) ax[s] = load_a_frag(rowb, s, lane);
      const unsigned short* wh = wpack + (size_t)hh * 128 * 8 * 512;
      #pragma unroll
      for (int t = 0; t < 16; ++t) {                 // wave owns 16 N-tiles
        v8f acc = {};
        const unsigned short* base = wh + (size_t)(wave * 16 + t) * 8 * 512;
        #pragma unroll
        for (int s = 0; s < 8; ++s) {
          Frag b = load_b_frag(base + s * 512, lane);
          acc = wmma_bf16(ax[s], b, acc);
        }
        const int col = (wave * 16 + t) * 16 + (lane & 15);
        const int rb  = (lane < 16) ? 0 : 8;
        #pragma unroll
        for (int v = 0; v < 8; ++v) F[(rb + v) * 2048 + col] = acc[v];
      }
    }
    __syncthreads();

    // ---- per-edge: gate partial dots + H = lrelu(Amsg[i]+Bmsg[j]+b) (bf16) ----
    {
      const int e = tid >> 2, q = tid & 3;
      const int ri = ((e >> 4) << 2) + ((e >> 2) & 3);
      const int rj = ((e >> 4) << 2) + (e & 3);
      const float* Fi = F + ri * 2048;
      const float* Fj = F + rj * 2048 + 512;
      float s = 0.f;
      for (int c = q * 128; c < q * 128 + 128; ++c)
        s += gw2s[c] * lrelu(Fi[c] + Fj[c] + gb1s[c]);
      part[e * 4 + q] = s;
    }
    for (int i = tid; i < 64 * 512; i += 256) {
      const int e = i >> 9, c = i & 511;
      const int ri = ((e >> 4) << 2) + ((e >> 2) & 3);
      const int rj = ((e >> 4) << 2) + (e & 3);
      H[i] = bf16rne(lrelu(F[ri * 2048 + 1024 + c] + F[rj * 2048 + 1536 + c] + mb1s[c]));
    }
    __syncthreads();

    if (tid < 64)
      gate[tid] = part[tid * 4] + part[tid * 4 + 1] + part[tid * 4 + 2] + part[tid * 4 + 3] + hp.gb2[0];
    __syncthreads();

    // ---- segment softmax (4 edges per self node), scaled by nbr_w^pow ----
    if (tid < 16) {
      const float pw = hp.pw[0];
      float g[4], m = -3.4e38f;
      for (int j = 0; j < 4; ++j) { g[j] = gate[tid * 4 + j]; m = fmaxf(m, g[j]); }
      const int cb = tid & ~3;
      float w[4], sum = 0.f;
      for (int j = 0; j < 4; ++j) {
        const float nw = ewt[nodeBase + cb + j];
        w[j] = __expf(pw * __logf(nw)) * __expf(g[j] - m);
        sum += w[j];
      }
      const float inv = 1.f / (sum + 1e-10f);
      for (int j = 0; j < 4; ++j) coef[tid * 4 + j] = w[j] * inv;
    }
    __syncthreads();   // F fully consumed -> MO may alias it

    // ---- msg GEMM: [64x512] @ [512x256] -> MO ----
    {
      const int mt = wave >> 1, ntb = (wave & 1) * 8;
      const unsigned short* rowb = H + (mt * 16 + (lane & 15)) * 512;
      Frag ah[16];
      #pragma unroll
      for (int s = 0; s < 16; ++s) ah[s] = load_a_frag(rowb, s, lane);
      const unsigned short* w2h = wm2 + (size_t)hh * 16 * 16 * 512;
      #pragma unroll
      for (int n = 0; n < 8; ++n) {
        const int nt = ntb + n;
        v8f acc = {};
        const unsigned short* base = w2h + (size_t)nt * 16 * 512;
        #pragma unroll
        for (int s = 0; s < 16; ++s) {
          Frag b = load_b_frag(base + s * 512, lane);
          acc = wmma_bf16(ah[s], b, acc);
        }
        const int col = nt * 16 + (lane & 15);
        const int rb  = mt * 16 + ((lane < 16) ? 0 : 8);
        #pragma unroll
        for (int v = 0; v < 8; ++v) MO[(rb + v) * 256 + col] = acc[v];
      }
    }
    __syncthreads();

    // ---- weighted segment-sum into pooled accumulator ----
    for (int i = tid; i < 16 * 256; i += 256) {
      const int nI = i >> 8, c = i & 255;
      float a = 0.f;
      for (int j = 0; j < 4; ++j)
        a += coef[nI * 4 + j] * (MO[(nI * 4 + j) * 256 + c] + mb2s[c]);
      pool[i] += a;
    }
    __syncthreads();
  }

  // residual: x += mean(heads); write f32 + bf16
  for (int i = tid; i < 16 * 256; i += 256) {
    const float xv = X[nodeBase * 256 + i] + pool[i] * (1.f / 3.f);
    X[nodeBase * 256 + i]  = xv;
    XB[nodeBase * 256 + i] = bf16rne(xv);
  }
}

// ---------------- crystal pooling ----------------
#define C_F     0        // float[16*1024]; first 16KB alias MO float[16*256]
#define C_XT    65536
#define C_H     73728    // ushort[16*512]
#define C_POOL  90112    // float[4*256]
#define C_GB1   94208
#define C_GW2   96256
#define C_MB1   98304
#define C_MB2   100352
#define C_PART  101376   // float[16*16]
#define C_GATE  102400
#define C_COEF  102464
#define C_TOTAL 102528

__global__ void __launch_bounds__(256) k_cry_pool(
    const unsigned short* __restrict__ XB,
    const unsigned short* __restrict__ wpack,   // [3][64][8] B-frags
    const unsigned short* __restrict__ wm2,     // [3][16][16] B-frags
    const float* __restrict__ ewt, HeadsArg ha, float* __restrict__ out) {
  extern __shared__ char smem[];
  float*          F    = (float*)(smem + C_F);
  float*          MO   = (float*)(smem + C_F);
  unsigned short* xt   = (unsigned short*)(smem + C_XT);
  unsigned short* H    = (unsigned short*)(smem + C_H);
  float*          pool = (float*)(smem + C_POOL);
  float*          gb1s = (float*)(smem + C_GB1);
  float*          gw2s = (float*)(smem + C_GW2);
  float*          mb1s = (float*)(smem + C_MB1);
  float*          mb2s = (float*)(smem + C_MB2);
  float*          part = (float*)(smem + C_PART);
  float*          gate = (float*)(smem + C_GATE);
  float*          coef = (float*)(smem + C_COEF);

  const int tid = threadIdx.x, lane = tid & 31, wave = tid >> 5;
  const int nodeBase = blockIdx.x * 16, cryBase = blockIdx.x * 4;

  for (int i = tid; i < 16 * 256; i += 256) xt[i] = XB[nodeBase * 256 + i];
  if (tid < 4 * 256) pool[tid] = 0.f;
  __syncthreads();

  for (int hh = 0; hh < 3; ++hh) {
    const HeadSmall hp = ha.h[hh];
    for (int i = tid; i < 512; i += 256) { gb1s[i] = hp.gb1[i]; gw2s[i] = hp.gw2[i]; mb1s[i] = hp.mb1[i]; }
    if (tid < 256) mb2s[tid] = hp.mb2[tid];
    __syncthreads();

    // fc1: [16x256] @ [256x1024] -> F = [gate_h | msg_h]
    {
      Frag ax[8];
      const unsigned short* rowb = xt + (lane & 15) * 256;
      #pragma unroll
      for (int s = 0; s < 8; ++s) ax[s] = load_a_frag(rowb, s, lane);
      const unsigned short* wh = wpack + (size_t)hh * 64 * 8 * 512;
      #pragma unroll
      for (int i = 0; i < 8; ++i) {
        const int t = wave * 8 + i;
        v8f acc = {};
        const unsigned short* base = wh + (size_t)t * 8 * 512;
        #pragma unroll
        for (int s = 0; s < 8; ++s) {
          Frag b = load_b_frag(base + s * 512, lane);
          acc = wmma_bf16(ax[s], b, acc);
        }
        const int col = t * 16 + (lane & 15);
        const int rb  = (lane < 16) ? 0 : 8;
        #pragma unroll
        for (int v = 0; v < 8; ++v) F[(rb + v) * 1024 + col] = acc[v];
      }
    }
    __syncthreads();

    {   // gate partial dots (16 nodes x 16 chunks of 32)
      const int r = tid >> 4, q = tid & 15;
      float s = 0.f;
      for (int c = q * 32; c < q * 32 + 32; ++c)
        s += gw2s[c] * lrelu(F[r * 1024 + c] + gb1s[c]);
      part[r * 16 + q] = s;
    }
    for (int i = tid; i < 16 * 512; i += 256) {
      const int r = i >> 9, c = i & 511;
      H[i] = bf16rne(lrelu(F[r * 1024 + 512 + c] + mb1s[c]));
    }
    __syncthreads();

    if (tid < 16) {
      float s = 0.f;
      for (int q = 0; q < 16; ++q) s += part[tid * 16 + q];
      gate[tid] = s + hp.gb2[0];
    }
    __syncthreads();

    if (tid < 4) {   // softmax over the 4 nodes of each crystal
      const float pw = hp.pw[0];
      float g[4], m = -3.4e38f;
      for (int j = 0; j < 4; ++j) { g[j] = gate[tid * 4 + j]; m = fmaxf(m, g[j]); }
      float w[4], sum = 0.f;
      for (int j = 0; j < 4; ++j) {
        const float nw = ewt[nodeBase + tid * 4 + j];
        w[j] = __expf(pw * __logf(nw)) * __expf(g[j] - m);
        sum += w[j];
      }
      const float inv = 1.f / (sum + 1e-10f);
      for (int j = 0; j < 4; ++j) coef[tid * 4 + j] = w[j] * inv;
    }
    __syncthreads();   // F consumed -> MO may alias

    // msg GEMM: [16x512] @ [512x256] -> MO
    {
      const unsigned short* rowb = H + (lane & 15) * 512;
      Frag ah[16];
      #pragma unroll
      for (int s = 0; s < 16; ++s) ah[s] = load_a_frag(rowb, s, lane);
      const unsigned short* w2h = wm2 + (size_t)hh * 16 * 16 * 512;
      #pragma unroll
      for (int n = 0; n < 2; ++n) {
        const int nt = wave * 2 + n;
        v8f acc = {};
        const unsigned short* base = w2h + (size_t)nt * 16 * 512;
        #pragma unroll
        for (int s = 0; s < 16; ++s) {
          Frag b = load_b_frag(base + s * 512, lane);
          acc = wmma_bf16(ah[s], b, acc);
        }
        const int col = nt * 16 + (lane & 15);
        const int rb  = (lane < 16) ? 0 : 8;
        #pragma unroll
        for (int v = 0; v < 8; ++v) MO[(rb + v) * 256 + col] = acc[v];
      }
    }
    __syncthreads();

    for (int i = tid; i < 4 * 256; i += 256) {
      const int cc = i >> 8, c = i & 255;
      float a = 0.f;
      for (int r = 0; r < 4; ++r)
        a += coef[cc * 4 + r] * (MO[(cc * 4 + r) * 256 + c] + mb2s[c]);
      pool[i] += a;
    }
    __syncthreads();
  }

  for (int i = tid; i < 4 * 256; i += 256)
    out[cryBase * 256 + i] = pool[i] * (1.f / 3.f);
}

// ---------------- host launcher ----------------
// d_in leaf order = jax.tree_util flatten (dict keys sorted at every level):
// 0 cry_elem_idx, 1 elem_fea, 2 elem_weights, 3 nbr_fea_idx,
// 4..30 cry_pool[0..2] (gate.fc1.W,b, gate.out.W,b, msg.fc1.W,b, msg.out.W,b, pow),
// 31 embed.W, 32 embed.b, 33..113 graphs[g][h] (same 9-leaf layout), 114 self_fea_idx.
extern "C" void kernel_launch(void* const* d_in, const int* in_sizes, int n_in,
                              void* d_out, int out_size, void* d_ws, size_t ws_size,
                              hipStream_t stream) {
  (void)in_sizes; (void)n_in; (void)out_size; (void)ws_size;
  auto F32 = [&](int i) { return (const float*)d_in[i]; };
  const float* ewt  = F32(2);
  const float* efea = F32(1);

  char* ws = (char*)d_ws;                       // ~54 MB of scratch used
  float*          X     = (float*)(ws + 0);                      // 33554432 B
  unsigned short* XB    = (unsigned short*)(ws + 33554432);      // 16777216 B
  unsigned short* WPACK = (unsigned short*)(ws + 50331648);      //  3145728 B
  unsigned short* WM2   = (unsigned short*)(ws + 53477376);      //   786432 B
  unsigned short* CPW   = (unsigned short*)(ws + 54263808);      //  1572864 B
  unsigned short* CPM2  = (unsigned short*)(ws + 55836672);      //   786432 B

  k_embed<<<NNODE, 256, 0, stream>>>(efea, ewt, F32(31), F32(32), X, XB);

  for (int g = 0; g < 3; ++g) {
    PackSrc ps; HeadsArg ha;
    for (int h = 0; h < 3; ++h) {
      const int b = 33 + 9 * (g * 3 + h);
      ps.gW[h] = F32(b + 0); ps.mW[h] = F32(b + 4); ps.m2W[h] = F32(b + 6);
      ha.h[h] = HeadSmall{F32(b + 1), F32(b + 2), F32(b + 3), F32(b + 5), F32(b + 7), F32(b + 8)};
    }
    k_pack_graph_fc1<<<98304 / 256, 256, 0, stream>>>(ps, WPACK);
    k_pack_m2<<<24576 / 256, 256, 0, stream>>>(ps, WM2);
    k_graph_layer<<<CRY / 4, 256, L_TOTAL, stream>>>(X, XB, WPACK, WM2, ewt, ha);
  }

  {
    PackSrc ps; HeadsArg ha;
    for (int h = 0; h < 3; ++h) {
      const int b = 4 + 9 * h;
      ps.gW[h] = F32(b + 0); ps.mW[h] = F32(b + 4); ps.m2W[h] = F32(b + 6);
      ha.h[h] = HeadSmall{F32(b + 1), F32(b + 2), F32(b + 3), F32(b + 5), F32(b + 7), F32(b + 8)};
    }
    k_pack_cry_fc1<<<49152 / 256, 256, 0, stream>>>(ps, CPW);
    k_pack_m2<<<24576 / 256, 256, 0, stream>>>(ps, CPM2);
    k_cry_pool<<<CRY / 4, 256, C_TOTAL, stream>>>(XB, CPW, CPM2, ewt, ha, (float*)d_out);
  }
}
